// Model_40827959116312
// MI455X (gfx1250) — matrix operand order repaired and checked by reading, not compiled, not run on hardware
//
#include <hip/hip_runtime.h>
#include <hip/hip_bf16.h>
#include <math.h>

typedef __attribute__((ext_vector_type(16))) _Float16 v16h;
typedef __attribute__((ext_vector_type(8)))  float    v8f;

#define DIN 20
#define DD  512
#define DH  256
#define KC  10

// ---------------- elementwise / utility kernels ----------------

__global__ void k_zero(float* p, long n) {
    long i = (long)blockIdx.x * blockDim.x + threadIdx.x;
    if (i < n) p[i] = 0.0f;
}

// h[n,d] = sum_k x[n,k] * Wi[k,d] + bi[d]   (K=20, plain VALU)
__global__ void k_init_linear(const float* __restrict__ x, const float* __restrict__ Wi,
                              const float* __restrict__ bi, float* __restrict__ h, long total) {
    long i = (long)blockIdx.x * blockDim.x + threadIdx.x;
    if (i >= total) return;
    int d = (int)(i % DD);
    long n = i / DD;
    float acc = bi[d];
    const float* xr = x + n * DIN;
#pragma unroll
    for (int k = 0; k < DIN; ++k) acc = fmaf(xr[k], Wi[(long)k * DD + d], acc);
    h[i] = acc;
}

__global__ void k_deg(const int* __restrict__ dst, float* __restrict__ deg, int E) {
    int e = blockIdx.x * blockDim.x + threadIdx.x;
    if (e < E) atomicAdd(&deg[dst[e]], 1.0f);
}

// per (edge, 4-float chunk): agg[dst] += h[src]
__global__ void k_scatter(const float* __restrict__ h, const int* __restrict__ src,
                          const int* __restrict__ dst, float* __restrict__ agg, long total) {
    long i = (long)blockIdx.x * blockDim.x + threadIdx.x;
    if (i >= total) return;
    long e  = i >> 7;          // D/4 = 128 chunks per edge
    int  c4 = (int)(i & 127) * 4;
    const float4 v = *(const float4*)(h + (long)src[e] * DD + c4);
    float* a = agg + (long)dst[e] * DD + c4;
    atomicAdd(a + 0, v.x); atomicAdd(a + 1, v.y);
    atomicAdd(a + 2, v.z); atomicAdd(a + 3, v.w);
}

// t = h + t / max(deg,1)
__global__ void k_combine(const float* __restrict__ h, float* __restrict__ t,
                          const float* __restrict__ deg, long total) {
    long i = (long)blockIdx.x * blockDim.x + threadIdx.x;
    if (i >= total) return;
    long n = i / DD;
    t[i] = h[i] + t[i] / fmaxf(deg[n], 1.0f);
}

// one block per column: mean + rsqrt(var+eps) over `rows`
__global__ void k_colstats(const float* __restrict__ t, int rows, int cols,
                           float* __restrict__ mean, float* __restrict__ inv) {
    __shared__ float ss[256], sq[256];
    int d = blockIdx.x, tid = threadIdx.x;
    float s = 0.f, q = 0.f;
    for (int r = tid; r < rows; r += 256) {
        float v = t[(long)r * cols + d];
        s += v; q += v * v;
    }
    ss[tid] = s; sq[tid] = q; __syncthreads();
    for (int st = 128; st > 0; st >>= 1) {
        if (tid < st) { ss[tid] += ss[tid + st]; sq[tid] += sq[tid + st]; }
        __syncthreads();
    }
    if (tid == 0) {
        float mu  = ss[0] / rows;
        float var = sq[0] / rows - mu * mu;
        mean[d] = mu;
        inv[d]  = rsqrtf(var + 1e-5f);
    }
}

// out = relu((in-mean)*inv*gamma+beta), optional f32 and f16 destinations
__global__ void k_bn_relu(const float* __restrict__ in, const float* __restrict__ mean,
                          const float* __restrict__ inv, const float* __restrict__ gamma,
                          const float* __restrict__ beta, float* __restrict__ out32,
                          _Float16* __restrict__ out16, long total, int cols) {
    long i = (long)blockIdx.x * blockDim.x + threadIdx.x;
    if (i >= total) return;
    int d = (int)(i % cols);
    float v = (in[i] - mean[d]) * inv[d] * gamma[d] + beta[d];
    v = fmaxf(v, 0.0f);
    if (out32) out32[i] = v;
    if (out16) out16[i] = (_Float16)v;
}

__global__ void k_qcnt(const int* __restrict__ n2g, float* __restrict__ qcnt, int N) {
    int n = blockIdx.x * blockDim.x + threadIdx.x;
    if (n < N) atomicAdd(&qcnt[n2g[n]], 1.0f);
}

__global__ void k_qsum(const float* __restrict__ h, const int* __restrict__ n2g,
                       float* __restrict__ qsum, long total) {
    long i = (long)blockIdx.x * blockDim.x + threadIdx.x;
    if (i >= total) return;
    long n = i / DD; int d = (int)(i % DD);
    atomicAdd(&qsum[(long)n2g[n] * DD + d], h[i]);
}

// build fp16 concat matrix b[row=g*KC+k, 0:3D] = [qemb | pg_emb | neigh_emb]
__global__ void k_build_b(const float* __restrict__ qsum, const float* __restrict__ qcnt,
                          const float* __restrict__ pg, const float* __restrict__ ne,
                          _Float16* __restrict__ bm, long total) {
    long i = (long)blockIdx.x * blockDim.x + threadIdx.x;
    if (i >= total) return;
    int  c   = (int)(i % (3 * DD));
    long row = i / (3 * DD);
    int  g   = (int)(row / KC);
    float v;
    if (c < DD)            v = qsum[(long)g * DD + c] / fmaxf(qcnt[g], 1.0f);
    else if (c < 2 * DD)   v = pg[(long)g * DD + (c - DD)];
    else                   v = ne[row * DD + (c - 2 * DD)];
    bm[i] = (_Float16)v;
}

// Convert W[Kk,Nn] (f32, row-major) into fp16 WMMA B-fragment tile order:
//   tile index  t  = nt*(Kk/32) + kt            (k-steps for a fixed nt are contiguous)
//   within tile: lane (0..31), element e (0..15)
//   value = W[kt*32 + (lane>>4)*16 + e][nt*16 + (lane&15)]
// So in the GEMM each lane reads its 16 B values as two aligned uint4 loads.
__global__ void k_pack_b(const float* __restrict__ W, _Float16* __restrict__ Bs,
                         int Kk, int Nn, long total) {
    long i = (long)blockIdx.x * blockDim.x + threadIdx.x;
    if (i >= total) return;
    int  within = (int)(i & 511);           // 512 halves per tile
    long tile   = i >> 9;
    int  lane = within >> 4, e = within & 15;
    int  g = lane >> 4, l = lane & 15;
    int  ktiles = Kk >> 5;
    int  nt = (int)(tile / ktiles), kt = (int)(tile % ktiles);
    int  k = kt * 32 + g * 16 + e;
    int  n = nt * 16 + l;
    Bs[i] = (_Float16)W[(long)k * Nn + n];
}

// ---------------- WMMA GEMM: C[M,Nn] = A[M,Kk] @ B[Kk,Nn] + bias ----------------
// One wave32 per 16(M) x 32(N) output: one A fragment feeds two WMMAs per k-step.
// A fragment (contiguous k order, matching k_pack_b): lane<16 holds row M=lane, K=k0+{0..15};
// lane>=16 holds row M=lane-16, K=k0+{16..31}  -> two aligned uint4 loads.
// B comes pre-swizzled by k_pack_b -> two aligned uint4 loads per fragment.
// C/D: VGPR r -> row m0 + r + 8*(lane>=16), col n0 + lane%16.
__global__ __launch_bounds__(32) void k_gemm_wmma(const _Float16* __restrict__ A,
                                                  const _Float16* __restrict__ Bs,
                                                  const float* __restrict__ bias,
                                                  float* __restrict__ C,
                                                  int M, int Nn, int Kk) {
    const int lane = threadIdx.x;
    const int g = lane >> 4, l = lane & 15;
    const int m0 = blockIdx.y * 16;
    const int nt0 = blockIdx.x * 2;                 // two 16-wide N tiles per wave
    const int ktiles = Kk >> 5;

    v8f acc0 = {}, acc1 = {};
    const _Float16* Arow = A + (long)(m0 + l) * Kk;
    const _Float16* B0 = Bs + ((long)nt0 * ktiles) * 512 + lane * 16;
    const _Float16* B1 = B0 + (long)ktiles * 512;

    for (int kt = 0; kt < ktiles; ++kt) {
        const int k0 = kt * 32;
        __builtin_prefetch(Arow + k0 + 128, 0, 0);   // global_prefetch_b8 ahead on A stream

        union { v16h v; uint4 q[2]; } af, bf0, bf1;
        af.q[0]  = *(const uint4*)(Arow + k0 + g * 16);
        af.q[1]  = *(const uint4*)(Arow + k0 + g * 16 + 8);
        bf0.q[0] = *(const uint4*)(B0 + (long)kt * 512);
        bf0.q[1] = *(const uint4*)(B0 + (long)kt * 512 + 8);
        bf1.q[0] = *(const uint4*)(B1 + (long)kt * 512);
        bf1.q[1] = *(const uint4*)(B1 + (long)kt * 512 + 8);

        acc0 = __builtin_amdgcn_wmma_f32_16x16x32_f16(false, af.v, false, bf0.v,
                                                      (short)0, acc0, false, false);
        acc1 = __builtin_amdgcn_wmma_f32_16x16x32_f16(false, af.v, false, bf1.v,
                                                      (short)0, acc1, false, false);
    }

    const int n0 = nt0 * 16;
    const float bv0 = bias ? bias[n0 + l]      : 0.0f;
    const float bv1 = bias ? bias[n0 + 16 + l] : 0.0f;
#pragma unroll
    for (int r = 0; r < 8; ++r) {
        const long row = (long)(m0 + g * 8 + r) * Nn;
        C[row + n0 + l]      = acc0[r] + bv0;
        C[row + n0 + 16 + l] = acc1[r] + bv1;
    }
}

// pred[row] = sigmoid(dot(H3[row,:], W4) + b4)
__global__ void k_final(const float* __restrict__ H, const float* __restrict__ W4,
                        const float* __restrict__ b4, float* __restrict__ out, int rows) {
    int r = blockIdx.x * blockDim.x + threadIdx.x;
    if (r >= rows) return;
    float acc = b4[0];
    const float* hr = H + (long)r * DH;
#pragma unroll 8
    for (int d = 0; d < DH; ++d) acc = fmaf(hr[d], W4[d], acc);
    out[r] = 1.0f / (1.0f + expf(-acc));
}

// ---------------- launch ----------------

extern "C" void kernel_launch(void* const* d_in, const int* in_sizes, int n_in,
                              void* d_out, int out_size, void* d_ws, size_t ws_size,
                              hipStream_t stream) {
    const float* x    = (const float*)d_in[0];
    const float* pg   = (const float*)d_in[1];
    const float* ne   = (const float*)d_in[2];
    const float* Wi   = (const float*)d_in[3];
    const float* bi   = (const float*)d_in[4];
    const float* g1   = (const float*)d_in[5];
    const float* be1  = (const float*)d_in[6];
    const float* g2   = (const float*)d_in[7];
    const float* be2  = (const float*)d_in[8];
    const float* Wfc  = (const float*)d_in[9];
    const float* bfc  = (const float*)d_in[10];
    const float* W2   = (const float*)d_in[11];
    const float* b2   = (const float*)d_in[12];
    const float* W3   = (const float*)d_in[13];
    const float* b3   = (const float*)d_in[14];
    const float* W4   = (const float*)d_in[15];
    const float* b4   = (const float*)d_in[16];
    const float* gb   = (const float*)d_in[17];
    const float* bb   = (const float*)d_in[18];
    const float* gb2  = (const float*)d_in[19];
    const float* bb2  = (const float*)d_in[20];
    const float* gb3  = (const float*)d_in[21];
    const float* bb3  = (const float*)d_in[22];
    const int*   esrc = (const int*)d_in[23];
    const int*   edst = (const int*)d_in[24];
    const int*   n2g  = (const int*)d_in[25];
    float* out = (float*)d_out;

    const int  N  = in_sizes[0] / DIN;       // 10000
    const int  E  = in_sizes[23];            // 160000
    const int  B  = in_sizes[1] / DD;        // 64
    const int  BK = B * KC;                  // 640
    const int  K3 = 3 * DD;                  // 1536
    const long ND = (long)N * DD;
    const long HD = (long)BK * DH;

    // carve workspace (256B aligned)
    char* w = (char*)d_ws; size_t off = 0;
    auto carve = [&](size_t bytes) -> void* {
        void* p = w + off; off += (bytes + 255) & ~(size_t)255; return p;
    };
    float*    bufA   = (float*)carve(ND * 4);
    float*    bufB   = (float*)carve(ND * 4);
    float*    deg    = (float*)carve((size_t)N * 4);
    float*    meanV  = (float*)carve(DD * 4);
    float*    invV   = (float*)carve(DD * 4);
    float*    qsum   = (float*)carve((size_t)B * DD * 4);
    float*    qcnt   = (float*)carve((size_t)B * 4);
    _Float16* bmat   = (_Float16*)carve((size_t)BK * K3 * 2);
    _Float16* Wfc_s  = (_Float16*)carve((size_t)K3 * DH * 2);   // swizzled fragments
    _Float16* W2_s   = (_Float16*)carve((size_t)DH * DH * 2);
    _Float16* W3_s   = (_Float16*)carve((size_t)DH * DH * 2);
    float*    Hbuf   = (float*)carve(HD * 4);
    _Float16* Hh     = (_Float16*)carve(HD * 2);
    (void)ws_size; (void)n_in; (void)out_size;

    const int T = 256;
    auto nb = [](long n, int t) { return (unsigned)((n + t - 1) / t); };

    // 1) initial linear
    k_init_linear<<<nb(ND, T), T, 0, stream>>>(x, Wi, bi, bufA, ND);

    // 2) degrees (same for both layers)
    k_zero<<<nb(N, T), T, 0, stream>>>(deg, N);
    k_deg<<<nb(E, T), T, 0, stream>>>(edst, deg, E);

    // 3) two GIN(mean) + BN + ReLU layers, h lives in bufA
    const float* gam[2] = { g1, g2 };
    const float* bet[2] = { be1, be2 };
    for (int layer = 0; layer < 2; ++layer) {
        k_zero<<<nb(ND, T), T, 0, stream>>>(bufB, ND);
        k_scatter<<<nb((long)E * (DD / 4), T), T, 0, stream>>>(bufA, esrc, edst, bufB, (long)E * (DD / 4));
        k_combine<<<nb(ND, T), T, 0, stream>>>(bufA, bufB, deg, ND);
        k_colstats<<<DD, 256, 0, stream>>>(bufB, N, DD, meanV, invV);
        k_bn_relu<<<nb(ND, T), T, 0, stream>>>(bufB, meanV, invV, gam[layer], bet[layer],
                                               bufA, (_Float16*)nullptr, ND, DD);
    }

    // 4) per-graph mean readout
    k_zero<<<nb((long)B * DD, T), T, 0, stream>>>(qsum, (long)B * DD);
    k_zero<<<nb(B, T), T, 0, stream>>>(qcnt, B);
    k_qcnt<<<nb(N, T), T, 0, stream>>>(n2g, qcnt, N);
    k_qsum<<<nb(ND, T), T, 0, stream>>>(bufA, n2g, qsum, ND);

    // 5) concat matrix (fp16) + fragment-swizzled fp16 weights
    k_build_b<<<nb((long)BK * K3, T), T, 0, stream>>>(qsum, qcnt, pg, ne, bmat, (long)BK * K3);
    k_pack_b<<<nb((long)K3 * DH, T), T, 0, stream>>>(Wfc, Wfc_s, K3, DH, (long)K3 * DH);
    k_pack_b<<<nb((long)DH * DH, T), T, 0, stream>>>(W2, W2_s, DH, DH, (long)DH * DH);
    k_pack_b<<<nb((long)DH * DH, T), T, 0, stream>>>(W3, W3_s, DH, DH, (long)DH * DH);

    // 6) MLP: three WMMA GEMMs with BN+ReLU between
    dim3 gridG(DH / 32, BK / 16);                    // 16x32 output tile per wave
    k_gemm_wmma<<<gridG, 32, 0, stream>>>(bmat, Wfc_s, bfc, Hbuf, BK, DH, K3);
    k_colstats<<<DH, 256, 0, stream>>>(Hbuf, BK, DH, meanV, invV);
    k_bn_relu<<<nb(HD, T), T, 0, stream>>>(Hbuf, meanV, invV, gb, bb, Hbuf, Hh, HD, DH);

    k_gemm_wmma<<<gridG, 32, 0, stream>>>(Hh, W2_s, b2, Hbuf, BK, DH, DH);
    k_colstats<<<DH, 256, 0, stream>>>(Hbuf, BK, DH, meanV, invV);
    k_bn_relu<<<nb(HD, T), T, 0, stream>>>(Hbuf, meanV, invV, gb2, bb2, Hbuf, Hh, HD, DH);

    k_gemm_wmma<<<gridG, 32, 0, stream>>>(Hh, W3_s, b3, Hbuf, BK, DH, DH);
    k_colstats<<<DH, 256, 0, stream>>>(Hbuf, BK, DH, meanV, invV);
    k_bn_relu<<<nb(HD, T), T, 0, stream>>>(Hbuf, meanV, invV, gb3, bb3, Hbuf, (_Float16*)nullptr, HD, DH);

    // 7) final projection + sigmoid -> [B, KC]
    k_final<<<nb(BK, T), T, 0, stream>>>(Hbuf, W4, b4, out, BK);
}
